// ContextAwareGating_82059645157924
// MI455X (gfx1250) — matrix operand order repaired and checked by reading, not compiled, not run on hardware
//
#include <hip/hip_runtime.h>
#include <math.h>

typedef _Float16 v16h __attribute__((ext_vector_type(16)));
typedef _Float16 v8h  __attribute__((ext_vector_type(8)));
typedef _Float16 v4h  __attribute__((ext_vector_type(4)));
typedef float    v8f  __attribute__((ext_vector_type(8)));
typedef float    v4f  __attribute__((ext_vector_type(4)));

constexpr int B    = 65536;
constexpr int D    = 1024;
constexpr int CDIM = 256;
constexpr int P    = 32;
constexpr int H    = 128;
constexpr int IDIM = 32;
constexpr int E    = 16;
constexpr int FUS  = 1056;   // D + P
constexpr int TM   = 16;     // rows per block
constexpr int NTHREADS = 256;

// padded LDS row strides (in elements) -> conflict-free b128 fragment loads
constexpr int FUS_S = 1064;  // halfs, 2128 B  (532 dw % 64 = 20 banks/lane step)
constexpr int CTX_S = 264;   // halfs, 528 B   (132 dw % 64 = 4)
constexpr int HF_S  = 136;   // halfs, 272 B   (68 dw % 64 = 4)
constexpr int G1_S  = 40;    // halfs, 80 B    (20 banks/lane step)
constexpr int IF_S  = 40;    // halfs
constexpr int SF    = 132;   // floats, 528 B

// f16 weight workspace layout (element offsets)
constexpr int OFF_W1  = 0;                 // gp1_w [FUS][H]
constexpr int OFF_WC1 = OFF_W1  + FUS * H; // cp1_w [CDIM][P]
constexpr int OFF_WC2 = OFF_WC1 + CDIM * P;// cp2_w [P][P]
constexpr int OFF_W2  = OFF_WC2 + P * P;   // gp2_w [H][IDIM]
constexpr int OFF_W3  = OFF_W2  + H * IDIM;// gp3_w [IDIM][E]
constexpr int WS_HALFS = OFF_W3 + IDIM * E;

__device__ __forceinline__ float gelu_erf(float v) {
  return 0.5f * v * (1.0f + erff(v * 0.70710678118654752f));
}

__device__ __forceinline__ v16h cat8(v8h lo, v8h hi) {
  return __builtin_shufflevector(lo, hi, 0, 1, 2, 3, 4, 5, 6, 7,
                                          8, 9, 10, 11, 12, 13, 14, 15);
}

// A operand (16x32 f16), ISA layout: lanes 0-15 row m=lane hold K {ks..+7, ks+16..+23},
// lanes 16-31 row m=lane-16 hold K {ks+8..+15, ks+24..+31}. Two aligned 16-byte runs.
__device__ __forceinline__ v16h load_a(const _Float16* buf, int rowStride, int ks, int lane) {
  const int m  = lane & 15;
  const int k0 = ks + ((lane & 16) ? 8 : 0);
  const _Float16* p = buf + m * rowStride + k0;
  v8h lo = *(const v8h*)(p);
  v8h hi = *(const v8h*)(p + 16);
  return cat8(lo, hi);
}

// B operand (32x16 f16): lane holds column n = n0 + lane%16, K in the same order as load_a.
__device__ __forceinline__ v16h load_b(const _Float16* w, int nStride, int ks, int n0, int lane) {
  const int n  = n0 + (lane & 15);
  const int k0 = ks + ((lane & 16) ? 8 : 0);
  v16h b;
#pragma unroll
  for (int s = 0; s < 8; ++s) {
    b[s]     = w[(k0 + s) * nStride + n];
    b[s + 8] = w[(k0 + 16 + s) * nStride + n];
  }
  return b;
}

// butterfly reduction over 16-lane groups (wave32: two rows per wave)
__device__ __forceinline__ void reduce16(float& s, float& s2) {
  for (int off = 8; off > 0; off >>= 1) {
    s  += __shfl_xor(s,  off, 16);
    s2 += __shfl_xor(s2, off, 16);
  }
}

__global__ void prep_weights(const float* __restrict__ gp1_w, const float* __restrict__ cp1_w,
                             const float* __restrict__ cp2_w, const float* __restrict__ gp2_w,
                             const float* __restrict__ gp3_w, _Float16* __restrict__ ws) {
  int i = blockIdx.x * blockDim.x + threadIdx.x;
  if (i < FUS * H)  { ws[OFF_W1  + i] = (_Float16)gp1_w[i]; return; }
  i -= FUS * H;
  if (i < CDIM * P) { ws[OFF_WC1 + i] = (_Float16)cp1_w[i]; return; }
  i -= CDIM * P;
  if (i < P * P)    { ws[OFF_WC2 + i] = (_Float16)cp2_w[i]; return; }
  i -= P * P;
  if (i < H * IDIM) { ws[OFF_W2  + i] = (_Float16)gp2_w[i]; return; }
  i -= H * IDIM;
  if (i < IDIM * E) { ws[OFF_W3  + i] = (_Float16)gp3_w[i]; return; }
}

__global__ void __launch_bounds__(NTHREADS)
gating_kernel(const float* __restrict__ x, const float* __restrict__ ctx_in,
              const float* __restrict__ en_g, const float* __restrict__ en_b,
              const float* __restrict__ cn_g, const float* __restrict__ cn_b,
              const float* __restrict__ cp1_b, const float* __restrict__ cp2_b,
              const float* __restrict__ cpn_g, const float* __restrict__ cpn_b,
              const float* __restrict__ fn_g, const float* __restrict__ fn_b,
              const float* __restrict__ gp1_b, const float* __restrict__ ln1_g,
              const float* __restrict__ ln1_b, const float* __restrict__ gp2_b,
              const float* __restrict__ ln2_g, const float* __restrict__ ln2_b,
              const float* __restrict__ gp3_b, const _Float16* __restrict__ ws,
              float* __restrict__ out_w, int* __restrict__ out_i,
              float* __restrict__ out_logits) {
  __shared__ _Float16 s_fusion[TM * FUS_S];  // raw x -> emb || cf -> normalized fusion
  __shared__ _Float16 s_ctx[TM * CTX_S];
  __shared__ _Float16 s_g1[TM * G1_S];
  __shared__ _Float16 s_hf[TM * HF_S];
  __shared__ _Float16 s_if[TM * IF_S];
  __shared__ float    s_f32[TM * SF];        // f32 staging for GEMM outputs / logits

  const int tid  = threadIdx.x;
  const int lane = tid & 31;
  const int wave = tid >> 5;
  const int r    = tid >> 4;   // row within tile (0..15)
  const int j    = tid & 15;   // 16 threads cooperate per row
  const int row0 = blockIdx.x * TM;

  const _Float16* W1  = ws + OFF_W1;
  const _Float16* WC1 = ws + OFF_WC1;
  const _Float16* WC2 = ws + OFF_WC2;
  const _Float16* W2  = ws + OFF_W2;
  const _Float16* W3  = ws + OFF_W3;

  // keep gp1 weights warm (L2-resident across all 4096 blocks)
  __builtin_prefetch(W1 + tid * (FUS * H / NTHREADS), 0, 0);

  // ---- Phase A: stream x once (NT b128 loads), stash raw f16 in LDS + f32 stats ----
  {
    const v4f* xr4 = (const v4f*)(x + (size_t)(row0 + r) * D);
    float s = 0.f, s2 = 0.f;
#pragma unroll 4
    for (int it = 0; it < D / 64; ++it) {           // 16 iters: 16 threads x 4 floats
      const int q = j + 16 * it;                    // float4 index within row
      v4f v = __builtin_nontemporal_load(&xr4[q]);
      s  += v[0] + v[1] + v[2] + v[3];
      s2 += v[0] * v[0] + v[1] * v[1] + v[2] * v[2] + v[3] * v[3];
      v4h h = { (_Float16)v[0], (_Float16)v[1], (_Float16)v[2], (_Float16)v[3] };
      *(v4h*)&s_fusion[r * FUS_S + q * 4] = h;
    }
    reduce16(s, s2);
    float mu = s * (1.0f / D);
    float rs = rsqrtf(s2 * (1.0f / D) - mu * mu + 1e-5f);
#pragma unroll 4
    for (int it = 0; it < D / 64; ++it) {           // normalize from LDS copy
      const int i0 = (j + 16 * it) * 4;
      v4h h = *(const v4h*)&s_fusion[r * FUS_S + i0];
      v4f g = *(const v4f*)&en_g[i0];
      v4f b = *(const v4f*)&en_b[i0];
      v4h o;
#pragma unroll
      for (int c = 0; c < 4; ++c)
        o[c] = (_Float16)(((float)h[c] - mu) * rs * g[c] + b[c]);
      *(v4h*)&s_fusion[r * FUS_S + i0] = o;
    }
  }

  // ---- Phase B1: ctx = LN(context), single NT streaming pass ----
  {
    const v4f* cr4 = (const v4f*)(ctx_in + (size_t)(row0 + r) * CDIM);
    float s = 0.f, s2 = 0.f;
#pragma unroll
    for (int it = 0; it < CDIM / 64; ++it) {        // 4 iters
      const int q = j + 16 * it;
      v4f v = __builtin_nontemporal_load(&cr4[q]);
      s  += v[0] + v[1] + v[2] + v[3];
      s2 += v[0] * v[0] + v[1] * v[1] + v[2] * v[2] + v[3] * v[3];
      v4h h = { (_Float16)v[0], (_Float16)v[1], (_Float16)v[2], (_Float16)v[3] };
      *(v4h*)&s_ctx[r * CTX_S + q * 4] = h;
    }
    reduce16(s, s2);
    float mu = s * (1.0f / CDIM);
    float rs = rsqrtf(s2 * (1.0f / CDIM) - mu * mu + 1e-5f);
#pragma unroll
    for (int it = 0; it < CDIM / 64; ++it) {
      const int i0 = (j + 16 * it) * 4;
      v4h h = *(const v4h*)&s_ctx[r * CTX_S + i0];
      v4f g = *(const v4f*)&cn_g[i0];
      v4f b = *(const v4f*)&cn_b[i0];
      v4h o;
#pragma unroll
      for (int c = 0; c < 4; ++c)
        o[c] = (_Float16)(((float)h[c] - mu) * rs * g[c] + b[c]);
      *(v4h*)&s_ctx[r * CTX_S + i0] = o;
    }
  }
  __syncthreads();

  // ---- Phase B2: g1 = gelu(ctx @ cp1_w + b)  [16x256]x[256x32], waves 0-1 ----
  if (wave < 2) {
    const int n0 = wave * 16;
    v8f acc = {};
    for (int ks = 0; ks < CDIM; ks += 32) {
      v16h a = load_a(s_ctx, CTX_S, ks, lane);
      v16h b = load_b(WC1, P, ks, n0, lane);
      acc = __builtin_amdgcn_wmma_f32_16x16x32_f16(false, a, false, b, (short)0, acc, false, false);
    }
    const int n  = n0 + (lane & 15);
    const int mh = (lane & 16) ? 8 : 0;
    const float bias = cp1_b[n];
#pragma unroll
    for (int v = 0; v < 8; ++v)
      s_g1[(v + mh) * G1_S + n] = (_Float16)gelu_erf(acc[v] + bias);
  }
  __syncthreads();

  // ---- Phase B3: cf_raw = g1 @ cp2_w + b  [16x32]x[32x32], waves 0-1 ----
  if (wave < 2) {
    const int n0 = wave * 16;
    v8f acc = {};
    v16h a = load_a(s_g1, G1_S, 0, lane);
    v16h b = load_b(WC2, P, 0, n0, lane);
    acc = __builtin_amdgcn_wmma_f32_16x16x32_f16(false, a, false, b, (short)0, acc, false, false);
    const int n  = n0 + (lane & 15);
    const int mh = (lane & 16) ? 8 : 0;
    const float bias = cp2_b[n];
#pragma unroll
    for (int v = 0; v < 8; ++v)
      s_f32[(v + mh) * SF + n] = acc[v] + bias;
  }
  __syncthreads();

  // ---- Phase B4: cf = LN(cf_raw) -> s_fusion[:, D:D+P] ----
  if (tid < TM) {
    float s = 0.f, s2 = 0.f;
    for (int i = 0; i < P; ++i) { float v = s_f32[tid * SF + i]; s += v; s2 += v * v; }
    float mu = s * (1.0f / P);
    float rs = rsqrtf(s2 * (1.0f / P) - mu * mu + 1e-5f);
    for (int i = 0; i < P; ++i)
      s_fusion[tid * FUS_S + D + i] =
          (_Float16)((s_f32[tid * SF + i] - mu) * rs * cpn_g[i] + cpn_b[i]);
  }
  __syncthreads();

  // ---- Phase C: fusion = LN(fusion) in place (vectorized, exclusive ownership) ----
  {
    float s = 0.f, s2 = 0.f;
#pragma unroll 4
    for (int it = 0; it < D / 64; ++it) {
      const int i0 = (j + 16 * it) * 4;
      v4h h = *(const v4h*)&s_fusion[r * FUS_S + i0];
#pragma unroll
      for (int c = 0; c < 4; ++c) { float v = (float)h[c]; s += v; s2 += v * v; }
    }
    if (j < 8) {                                    // tail: D .. D+P-1 (32 halfs)
      v4h h = *(const v4h*)&s_fusion[r * FUS_S + D + j * 4];
#pragma unroll
      for (int c = 0; c < 4; ++c) { float v = (float)h[c]; s += v; s2 += v * v; }
    }
    reduce16(s, s2);
    float mu = s * (1.0f / FUS);
    float rs = rsqrtf(s2 * (1.0f / FUS) - mu * mu + 1e-5f);
#pragma unroll 4
    for (int it = 0; it < D / 64; ++it) {
      const int i0 = (j + 16 * it) * 4;
      v4h h = *(const v4h*)&s_fusion[r * FUS_S + i0];
      v4f g = *(const v4f*)&fn_g[i0];
      v4f b = *(const v4f*)&fn_b[i0];
      v4h o;
#pragma unroll
      for (int c = 0; c < 4; ++c)
        o[c] = (_Float16)(((float)h[c] - mu) * rs * g[c] + b[c]);
      *(v4h*)&s_fusion[r * FUS_S + i0] = o;
    }
    if (j < 8) {
      const int i0 = D + j * 4;
      v4h h = *(const v4h*)&s_fusion[r * FUS_S + i0];
      v4f g = *(const v4f*)&fn_g[i0];
      v4f b = *(const v4f*)&fn_b[i0];
      v4h o;
#pragma unroll
      for (int c = 0; c < 4; ++c)
        o[c] = (_Float16)(((float)h[c] - mu) * rs * g[c] + b[c]);
      *(v4h*)&s_fusion[r * FUS_S + i0] = o;
    }
  }
  __syncthreads();

  // ---- Phase D: h_raw = fusion @ gp1_w + b  [16x1056]x[1056x128], all 8 waves ----
  {
    const int n0 = wave * 16;
    v8f acc = {};
    for (int ks = 0; ks < FUS; ks += 32) {
      v16h a = load_a(s_fusion, FUS_S, ks, lane);
      v16h b = load_b(W1, H, ks, n0, lane);
      acc = __builtin_amdgcn_wmma_f32_16x16x32_f16(false, a, false, b, (short)0, acc, false, false);
    }
    const int n  = n0 + (lane & 15);
    const int mh = (lane & 16) ? 8 : 0;
    const float bias = gp1_b[n];
#pragma unroll
    for (int v = 0; v < 8; ++v)
      s_f32[(v + mh) * SF + n] = acc[v] + bias;
  }
  __syncthreads();

  // ---- Phase D2: h = gelu(LN(h_raw)) -> s_hf ----
  {
    float s = 0.f, s2 = 0.f;
#pragma unroll
    for (int it = 0; it < H / 64; ++it) {           // 2 iters of float4
      const int i0 = (j + 16 * it) * 4;
      v4f v = *(const v4f*)&s_f32[r * SF + i0];
#pragma unroll
      for (int c = 0; c < 4; ++c) { s += v[c]; s2 += v[c] * v[c]; }
    }
    reduce16(s, s2);
    float mu = s * (1.0f / H);
    float rs = rsqrtf(s2 * (1.0f / H) - mu * mu + 1e-5f);
#pragma unroll
    for (int it = 0; it < H / 64; ++it) {
      const int i0 = (j + 16 * it) * 4;
      v4f v = *(const v4f*)&s_f32[r * SF + i0];
      v4f g = *(const v4f*)&ln1_g[i0];
      v4f b = *(const v4f*)&ln1_b[i0];
      v4h o;
#pragma unroll
      for (int c = 0; c < 4; ++c)
        o[c] = (_Float16)gelu_erf((v[c] - mu) * rs * g[c] + b[c]);
      *(v4h*)&s_hf[r * HF_S + i0] = o;
    }
  }
  __syncthreads();

  // ---- Phase E: inter_raw = h @ gp2_w + b  [16x128]x[128x32], waves 0-1 ----
  if (wave < 2) {
    const int n0 = wave * 16;
    v8f acc = {};
    for (int ks = 0; ks < H; ks += 32) {
      v16h a = load_a(s_hf, HF_S, ks, lane);
      v16h b = load_b(W2, IDIM, ks, n0, lane);
      acc = __builtin_amdgcn_wmma_f32_16x16x32_f16(false, a, false, b, (short)0, acc, false, false);
    }
    const int n  = n0 + (lane & 15);
    const int mh = (lane & 16) ? 8 : 0;
    const float bias = gp2_b[n];
#pragma unroll
    for (int v = 0; v < 8; ++v)
      s_f32[(v + mh) * SF + n] = acc[v] + bias;
  }
  __syncthreads();

  // ---- Phase E2: inter = gelu(LN(inter_raw)) -> s_if ----
  if (tid < TM) {
    float s = 0.f, s2 = 0.f;
    for (int i = 0; i < IDIM; ++i) { float v = s_f32[tid * SF + i]; s += v; s2 += v * v; }
    float mu = s * (1.0f / IDIM);
    float rs = rsqrtf(s2 * (1.0f / IDIM) - mu * mu + 1e-5f);
    for (int i = 0; i < IDIM; ++i)
      s_if[tid * IF_S + i] =
          (_Float16)gelu_erf((s_f32[tid * SF + i] - mu) * rs * ln2_g[i] + ln2_b[i]);
  }
  __syncthreads();

  // ---- Phase F: logits = inter @ gp3_w + b  [16x32]x[32x16], wave 0 ----
  if (wave == 0) {
    v8f acc = {};
    v16h a = load_a(s_if, IF_S, 0, lane);
    v16h b = load_b(W3, E, 0, 0, lane);
    acc = __builtin_amdgcn_wmma_f32_16x16x32_f16(false, a, false, b, (short)0, acc, false, false);
    const int n  = lane & 15;
    const int mh = (lane & 16) ? 8 : 0;
    const float bias = gp3_b[n];
#pragma unroll
    for (int v = 0; v < 8; ++v)
      s_f32[(v + mh) * SF + n] = acc[v] + bias;
  }
  __syncthreads();

  // ---- Phase G: emit logits (NT b128), top-2 (first-index ties), 2-way softmax ----
  if (tid < TM) {
    const int grow = row0 + tid;
    float v0 = -3.4e38f, v1 = -3.4e38f;
    int   i0 = 0,        i1 = 0;
#pragma unroll
    for (int q = 0; q < E / 4; ++q) {
      v4f v = *(const v4f*)&s_f32[tid * SF + q * 4];
      __builtin_nontemporal_store(v, (v4f*)&out_logits[(size_t)grow * E + q * 4]);
#pragma unroll
      for (int c = 0; c < 4; ++c) {
        const int e = q * 4 + c;
        if (v[c] > v0)      { v1 = v0; i1 = i0; v0 = v[c]; i0 = e; }
        else if (v[c] > v1) { v1 = v[c]; i1 = e; }
      }
    }
    float e1    = expf(v1 - v0);
    float denom = 1.0f + e1;
    out_w[grow * 2 + 0] = 1.0f / denom;
    out_w[grow * 2 + 1] = e1 / denom;
    out_i[grow * 2 + 0] = i0;
    out_i[grow * 2 + 1] = i1;
  }
}

extern "C" void kernel_launch(void* const* d_in, const int* in_sizes, int n_in,
                              void* d_out, int out_size, void* d_ws, size_t ws_size,
                              hipStream_t stream) {
  const float* x       = (const float*)d_in[0];
  const float* context = (const float*)d_in[1];
  const float* en_g  = (const float*)d_in[2];
  const float* en_b  = (const float*)d_in[3];
  const float* cn_g  = (const float*)d_in[4];
  const float* cn_b  = (const float*)d_in[5];
  const float* cp1_w = (const float*)d_in[6];
  const float* cp1_b = (const float*)d_in[7];
  const float* cp2_w = (const float*)d_in[8];
  const float* cp2_b = (const float*)d_in[9];
  const float* cpn_g = (const float*)d_in[10];
  const float* cpn_b = (const float*)d_in[11];
  const float* fn_g  = (const float*)d_in[12];
  const float* fn_b  = (const float*)d_in[13];
  const float* gp1_w = (const float*)d_in[14];
  const float* gp1_b = (const float*)d_in[15];
  const float* ln1_g = (const float*)d_in[16];
  const float* ln1_b = (const float*)d_in[17];
  const float* gp2_w = (const float*)d_in[18];
  const float* gp2_b = (const float*)d_in[19];
  const float* ln2_g = (const float*)d_in[20];
  const float* ln2_b = (const float*)d_in[21];
  const float* gp3_w = (const float*)d_in[22];
  const float* gp3_b = (const float*)d_in[23];

  _Float16* ws = (_Float16*)d_ws;
  float* outf       = (float*)d_out;
  float* out_w      = outf;                         // [B,2] f32
  int*   out_i      = (int*)(outf + (size_t)B * 2); // [B,2] i32
  float* out_logits = outf + (size_t)B * 4;         // [B,16] f32

  prep_weights<<<(WS_HALFS + 255) / 256, 256, 0, stream>>>(gp1_w, cp1_w, cp2_w, gp2_w, gp3_w, ws);
  gating_kernel<<<B / TM, NTHREADS, 0, stream>>>(
      x, context, en_g, en_b, cn_g, cn_b, cp1_b, cp2_b, cpn_g, cpn_b, fn_g, fn_b,
      gp1_b, ln1_g, ln1_b, gp2_b, ln2_g, ln2_b, gp3_b, ws, out_w, out_i, out_logits);
}
